// JittableSelfAttention_Rels_89558658056630
// MI455X (gfx1250) — compile-verified, matrix operand order checked by reading, not run on hardware
//
#include <hip/hip_runtime.h>

// ---------------------------------------------------------------------------
// Temporal graph attention (TGAT-style) forward, algebraically reduced.
// softmax over the singleton query dim == 1, so:
//   out = sum_masked v_rows + (sum_masked te_rows) @ Wt_v + (sum_masked rels) @ We_v
// Kernel 1: 16 blocks do the masked gather-reductions (v gather, Time2Vec sin,
//           rels), partials to d_ws.  Kernel 2: one wave32 combines partials and
//           does the [1,64]@[64,128] + [1,32]@[32,128] projection with fp32 WMMA.
// ---------------------------------------------------------------------------

#define E_TOTAL 2048
#define HIDDEN  128
#define T_DIM   64
#define R_DIM   32
#define NBLK    16
#define EPB     (E_TOTAL / NBLK)           // 128 edges per block
#define PART    (HIDDEN + T_DIM + R_DIM)   // 224 floats of partials per block

typedef float    v2f  __attribute__((ext_vector_type(2)));
typedef float    v8f  __attribute__((ext_vector_type(8)));
typedef _Float16 v16h __attribute__((ext_vector_type(16)));

// ------------------------------ kernel 1 -----------------------------------
__global__ void __launch_bounds__(256)
tga_gather_reduce(const float* __restrict__ v_,
                  const int* __restrict__ neighbors,
                  const unsigned char* __restrict__ mask,   // jnp bool_: 1 byte
                  const float* __restrict__ times,
                  const float* __restrict__ rels,
                  const float* __restrict__ t2v_w0,
                  const float* __restrict__ t2v_b0,
                  const float* __restrict__ t2v_w,
                  const float* __restrict__ t2v_b,
                  float* __restrict__ ws) {
  __shared__ float acc[PART];   // [0,128): sum_v, [128,192): sum_te, [192,224): sum_rel
  const int t = threadIdx.x;
  if (t < PART) acc[t] = 0.f;
  __syncthreads();
  const int e0 = blockIdx.x * EPB;

  // masked sum of gathered V rows (coalesced: 128 lanes cover one 512B row)
  {
    const int j   = t & (HIDDEN - 1);
    const int sub = t >> 7;                       // 0..1
    float a = 0.f;
    for (int el = sub; el < EPB; el += 2) {
      const int e = e0 + el;
      if (mask[e]) a += v_[(size_t)neighbors[e] * HIDDEN + j];
    }
    atomicAdd(&acc[j], a);                        // ds_add_f32
  }
  // masked sum of Time2Vec rows te[e+1] = [t*w0+b0, sin(t*w + b)]
  {
    const int c   = t & (T_DIM - 1);
    const int sub = t >> 6;                       // 0..3
    const float wc = (c == 0) ? t2v_w0[0] : t2v_w[c - 1];
    const float bc = (c == 0) ? t2v_b0[0] : t2v_b[c - 1];
    float a = 0.f;
    for (int el = sub; el < EPB; el += 4) {
      const int e = e0 + el;
      if (mask[e]) {
        const float x = times[e] * wc + bc;
        a += (c == 0) ? x : __sinf(x);            // v_sin_f32
      }
    }
    atomicAdd(&acc[HIDDEN + c], a);
  }
  // masked sum of rel rows
  {
    const int c   = t & (R_DIM - 1);
    const int sub = t >> 5;                       // 0..7
    float a = 0.f;
    for (int el = sub; el < EPB; el += 8) {
      const int e = e0 + el;
      if (mask[e]) a += rels[e * R_DIM + c];
    }
    atomicAdd(&acc[HIDDEN + T_DIM + c], a);
  }
  __syncthreads();
  if (t < PART) ws[blockIdx.x * PART + t] = acc[t];
}

// ------------------------------ kernel 2 -----------------------------------
// One wave32.  Combines the NBLK partials, then:
//   out[0:128] = sv + ste @ Wt_v + srel @ We_v     (V-slice = columns 256..383)
// Projection done with WMMA: the 1x64 / 1x32 vector sits in row 0 of a 16xK
// A-operand; row 0 of the 16x16 f32 D accumulator (VGPR0, lanes 0-15) is the
// result for each 16-column tile.
__global__ void __launch_bounds__(32)
tga_project(const float* __restrict__ ws,
            const float* __restrict__ timeW,   // [T_DIM, 3*HIDDEN] row-major
            const float* __restrict__ edgeW,   // [R_DIM, 3*HIDDEN] row-major
            float* __restrict__ out) {
  __shared__ float sv[HIDDEN];
  __shared__ float ste[T_DIM];
  __shared__ float srel[R_DIM];
  const int lane = threadIdx.x;

  __builtin_prefetch(timeW + 2 * HIDDEN, 0, 0);   // global_prefetch_b8
  __builtin_prefetch(edgeW + 2 * HIDDEN, 0, 0);

  for (int j = lane; j < PART; j += 32) {
    float s = 0.f;
#pragma unroll
    for (int b = 0; b < NBLK; ++b) s += ws[b * PART + j];
    if (j < HIDDEN)              sv[j] = s;
    else if (j < HIDDEN + T_DIM) ste[j - HIDDEN] = s;
    else                         srel[j - HIDDEN - T_DIM] = s;
  }
  __syncthreads();

  const int half = lane >> 4;   // A/B: lanes 16-31 carry the +2 (f32) K slice
  const int nl   = lane & 15;   // column within a 16-wide tile

#if defined(__has_builtin) && __has_builtin(__builtin_amdgcn_wmma_f32_16x16x4_f32)
  // ---------------- fp32 WMMA path (full precision) ----------------
  // A 16x4 f32 layout: lanes 0-15 hold M=0..15 {K0,K1}; lanes 16-31 {K2,K3}.
  // Only row 0 (lanes 0 and 16) carries data; rows 1..15 are zero.
  v2f aT[T_DIM / 4];
#pragma unroll
  for (int kk = 0; kk < T_DIM / 4; ++kk) {
    const int kb = kk * 4 + half * 2;
    float ax = 0.f, ay = 0.f;
    if (nl == 0) { ax = ste[kb]; ay = ste[kb + 1]; }
    aT[kk].x = ax; aT[kk].y = ay;
  }
  v2f aE[R_DIM / 4];
#pragma unroll
  for (int kk = 0; kk < R_DIM / 4; ++kk) {
    const int kb = kk * 4 + half * 2;
    float ax = 0.f, ay = 0.f;
    if (nl == 0) { ax = srel[kb]; ay = srel[kb + 1]; }
    aE[kk].x = ax; aE[kk].y = ay;
  }

  for (int tile = 0; tile < HIDDEN / 16; ++tile) {
    const int jc = tile * 16;
    v8f c = {0.f, 0.f, 0.f, 0.f, 0.f, 0.f, 0.f, 0.f};
#pragma unroll
    for (int kk = 0; kk < T_DIM / 4; ++kk) {      // K = 64 over the time proj
      const int kb = kk * 4 + half * 2;
      v2f b;
      b.x = timeW[(size_t)kb       * (3 * HIDDEN) + 2 * HIDDEN + jc + nl];
      b.y = timeW[(size_t)(kb + 1) * (3 * HIDDEN) + 2 * HIDDEN + jc + nl];
      c = __builtin_amdgcn_wmma_f32_16x16x4_f32(false, aT[kk], false, b,
                                                (short)0, c, false, false);
    }
#pragma unroll
    for (int kk = 0; kk < R_DIM / 4; ++kk) {      // K = 32 over the edge proj
      const int kb = kk * 4 + half * 2;
      v2f b;
      b.x = edgeW[(size_t)kb       * (3 * HIDDEN) + 2 * HIDDEN + jc + nl];
      b.y = edgeW[(size_t)(kb + 1) * (3 * HIDDEN) + 2 * HIDDEN + jc + nl];
      c = __builtin_amdgcn_wmma_f32_16x16x4_f32(false, aE[kk], false, b,
                                                (short)0, c, false, false);
    }
    if (lane < 16) out[jc + lane] = c[0] + sv[jc + lane];   // D row 0 = VGPR0
  }
#else
  // ---------------- f16 WMMA fallback (codegen-confirmed builtin) ----------
  // A 16x32 f16 layout: lanes 0-15: v16h elems 0-7 -> K0-7, 8-15 -> K16-23;
  // lanes 16-31: +8.  B 32x16: elem i -> K = 16*half + i, col = nl.
  v16h aT[2], aE;
#pragma unroll
  for (int i = 0; i < 16; ++i) {
    const int kloc = half * 8 + i + ((i & 8) ? 8 : 0);      // 0..31
    aT[0][i] = (nl == 0) ? (_Float16)ste[kloc]       : (_Float16)0.f;
    aT[1][i] = (nl == 0) ? (_Float16)ste[kloc + 32]  : (_Float16)0.f;
    aE[i]    = (nl == 0) ? (_Float16)srel[kloc]      : (_Float16)0.f;
  }
  for (int tile = 0; tile < HIDDEN / 16; ++tile) {
    const int jc = tile * 16;
    v8f c = {0.f, 0.f, 0.f, 0.f, 0.f, 0.f, 0.f, 0.f};
#pragma unroll
    for (int ks = 0; ks < 2; ++ks) {
      v16h b;
#pragma unroll
      for (int i = 0; i < 16; ++i) {
        const int kb = ks * 32 + half * 16 + i;
        b[i] = (_Float16)timeW[(size_t)kb * (3 * HIDDEN) + 2 * HIDDEN + jc + nl];
      }
      c = __builtin_amdgcn_wmma_f32_16x16x32_f16(false, aT[ks], false, b,
                                                 (short)0, c, false, false);
    }
    {
      v16h b;
#pragma unroll
      for (int i = 0; i < 16; ++i) {
        const int kb = half * 16 + i;
        b[i] = (_Float16)edgeW[(size_t)kb * (3 * HIDDEN) + 2 * HIDDEN + jc + nl];
      }
      c = __builtin_amdgcn_wmma_f32_16x16x32_f16(false, aE, false, b,
                                                 (short)0, c, false, false);
    }
    if (lane < 16) out[jc + lane] = c[0] + sv[jc + lane];
  }
#endif
}

// ------------------------------ launcher -----------------------------------
extern "C" void kernel_launch(void* const* d_in, const int* in_sizes, int n_in,
                              void* d_out, int out_size, void* d_ws, size_t ws_size,
                              hipStream_t stream) {
  (void)in_sizes; (void)n_in; (void)out_size; (void)ws_size;
  // setup_inputs order:
  // 0 k_ (dead), 1 q_ (dead), 2 v_, 3 neighbors, 4 nid (dead), 5 mask,
  // 6 start_t (dead: feeds q only), 7 times, 8 rels, 9 t2v_w0, 10 t2v_b0,
  // 11 t2v_w, 12 t2v_b, 13 time_kqv_w, 14 edge_kqv_w
  const float*         v_        = (const float*)d_in[2];
  const int*           neighbors = (const int*)d_in[3];
  const unsigned char* mask      = (const unsigned char*)d_in[5];
  const float*         times     = (const float*)d_in[7];
  const float*         rels      = (const float*)d_in[8];
  const float*         t2v_w0    = (const float*)d_in[9];
  const float*         t2v_b0    = (const float*)d_in[10];
  const float*         t2v_w     = (const float*)d_in[11];
  const float*         t2v_b     = (const float*)d_in[12];
  const float*         timeW     = (const float*)d_in[13];
  const float*         edgeW     = (const float*)d_in[14];
  float* out = (float*)d_out;
  float* ws  = (float*)d_ws;    // needs NBLK*PART*4 = 14336 bytes

  tga_gather_reduce<<<dim3(NBLK), dim3(256), 0, stream>>>(
      v_, neighbors, mask, times, rels, t2v_w0, t2v_b0, t2v_w, t2v_b, ws);
  tga_project<<<dim3(1), dim3(32), 0, stream>>>(ws, timeW, edgeW, out);
}
